// MHAttention_37203006718687
// MI455X (gfx1250) — matrix operand order repaired and checked by reading, not compiled, not run on hardware
//
#include <hip/hip_runtime.h>

// MI455X / gfx1250 flash-attention for chunked + sliding-window mask.
// wave32, V_WMMA_F32_16X16X32_F16 (K=32 == head_dim), fp32 accumulate.
// Double-buffered P staging -> one s_barrier per key-block iteration.

typedef __attribute__((ext_vector_type(16))) _Float16 v16h;
typedef __attribute__((ext_vector_type(8)))  _Float16 v8h;
typedef __attribute__((ext_vector_type(8)))  float    v8f;

#define NHEADS 16
#define DHEAD  32
#define SEQ    2048
#define EMB    512
#define QCHUNK 64
#define WINDOW 256
#define MAXKEY 320      // 64 (own chunk) + 256 (window)
#define KROW   40       // K row pad: 80B rows -> 16B aligned frags, conflict-free
#define VROW   328      // V^T row pad: 656B rows -> 16B aligned frags, conflict-free
#define NEGBIG (-1.0e30f)

union F16x16 { v16h v; v8h h[2]; };

__device__ __forceinline__ float redmax16(float x) {
#pragma unroll
  for (int o = 1; o < 16; o <<= 1) x = fmaxf(x, __shfl_xor(x, o, 16));
  return x;
}
__device__ __forceinline__ float redsum16(float x) {
#pragma unroll
  for (int o = 1; o < 16; o <<= 1) x += __shfl_xor(x, o, 16);
  return x;
}

__global__ __launch_bounds__(128)
void mha_chunk_window_kernel(const float* __restrict__ q,
                             const float* __restrict__ k,
                             const float* __restrict__ v,
                             float* __restrict__ out) {
  const int qc   = blockIdx.x;   // query chunk 0..31
  const int head = blockIdx.y;   // 0..15
  const int b    = blockIdx.z;   // 0..3
  const int tid  = threadIdx.x;  // 0..127
  const int w    = tid >> 5;     // wave id 0..3 -> 16-query subtile
  const int lane = tid & 31;
  const int l15  = lane & 15;
  const int half = lane >> 4;

  __shared__ __align__(16) _Float16 Klds[MAXKEY][KROW];   // [key][d], f16
  __shared__ __align__(16) _Float16 VldsT[DHEAD][VROW];   // [d][key], f16
  __shared__ __align__(16) _Float16 Pst[4][2][16][32];    // per-wave, double-buffered

  const int qstart = qc * QCHUNK;
  const int kstart = (qstart >= WINDOW) ? (qstart - WINDOW) : 0;
  const int nk     = qstart + QCHUNK - kstart;            // multiple of 64, <=320

  // ---- Cooperative load: K -> Klds (f16), V -> VldsT (f16, transposed) ----
  {
    const size_t base = ((size_t)b * SEQ + (size_t)kstart) * EMB + (size_t)head * DHEAD;
    const int total = nk * 8;                             // float4 jobs per matrix
    for (int idx = tid; idx < total; idx += 128) {
      const int kl = idx >> 3;
      const int dv = (idx & 7) * 4;
      const float4 kf = *(const float4*)(k + base + (size_t)kl * EMB + dv);
      Klds[kl][dv + 0] = (_Float16)kf.x;
      Klds[kl][dv + 1] = (_Float16)kf.y;
      Klds[kl][dv + 2] = (_Float16)kf.z;
      Klds[kl][dv + 3] = (_Float16)kf.w;
      const float4 vf = *(const float4*)(v + base + (size_t)kl * EMB + dv);
      VldsT[dv + 0][kl] = (_Float16)vf.x;
      VldsT[dv + 1][kl] = (_Float16)vf.y;
      VldsT[dv + 2][kl] = (_Float16)vf.z;
      VldsT[dv + 3][kl] = (_Float16)vf.w;
    }
  }
  __syncthreads();

  // ---- Q A-fragment (16x32): lane row M=l15; elems j -> d=half*16+j
  //      (contiguous k order, matching B fragments) ----
  F16x16 aQ;
  {
    const int qrow = qstart + w * 16 + l15;
    const float* qp = q + ((size_t)b * SEQ + qrow) * EMB + (size_t)head * DHEAD;
    const float4 f0 = *(const float4*)(qp + half * 16);
    const float4 f1 = *(const float4*)(qp + half * 16 + 4);
    const float4 f2 = *(const float4*)(qp + half * 16 + 8);
    const float4 f3 = *(const float4*)(qp + half * 16 + 12);
    aQ.v[0]  = (_Float16)f0.x; aQ.v[1]  = (_Float16)f0.y;
    aQ.v[2]  = (_Float16)f0.z; aQ.v[3]  = (_Float16)f0.w;
    aQ.v[4]  = (_Float16)f1.x; aQ.v[5]  = (_Float16)f1.y;
    aQ.v[6]  = (_Float16)f1.z; aQ.v[7]  = (_Float16)f1.w;
    aQ.v[8]  = (_Float16)f2.x; aQ.v[9]  = (_Float16)f2.y;
    aQ.v[10] = (_Float16)f2.z; aQ.v[11] = (_Float16)f2.w;
    aQ.v[12] = (_Float16)f3.x; aQ.v[13] = (_Float16)f3.y;
    aQ.v[14] = (_Float16)f3.z; aQ.v[15] = (_Float16)f3.w;
  }

  // Online-softmax state: this lane owns rows M = half*8 + r (r=0..7),
  // matching the f32 C/D VGPR layout.
  v8f acc0 = {};  // out cols d = l15
  v8f acc1 = {};  // out cols d = 16 + l15
  float mrow[8], lrow[8];
#pragma unroll
  for (int r = 0; r < 8; ++r) { mrow[r] = NEGBIG; lrow[r] = 0.0f; }

  const int   qbase = qstart + w * 16 + half * 8;
  const float scale = 0.17677669529663687f;  // 1/sqrt(32)
  const int   nblk  = nk >> 5;               // 32-key blocks
  const v8f   zero  = {};

  for (int kb = 0; kb < nblk; ++kb) {
    const int kb0 = kb * 32;
    const int buf = kb & 1;

    // K^T B-fragments (lane = key column, elems = 16 consecutive d values)
    F16x16 bk0, bk1;
    { const v8h* p = (const v8h*)&Klds[kb0 + l15][half * 16];      bk0.h[0] = p[0]; bk0.h[1] = p[1]; }
    { const v8h* p = (const v8h*)&Klds[kb0 + 16 + l15][half * 16]; bk1.h[0] = p[0]; bk1.h[1] = p[1]; }

    v8f s0 = __builtin_amdgcn_wmma_f32_16x16x32_f16(false, aQ.v, false, bk0.v, (short)0, zero, false, false);
    v8f s1 = __builtin_amdgcn_wmma_f32_16x16x32_f16(false, aQ.v, false, bk1.v, (short)0, zero, false, false);

    // Mask + online softmax in D layout; allowed <=> kg >= qg - (WINDOW-1)
    const int kg0 = kstart + kb0 + l15;
    const int kg1 = kg0 + 16;
    float corr[8];
#pragma unroll
    for (int r = 0; r < 8; ++r) {
      const int qg = qbase + r;
      float x0 = (kg0 + (WINDOW - 1) >= qg) ? s0[r] * scale : NEGBIG;
      float x1 = (kg1 + (WINDOW - 1) >= qg) ? s1[r] * scale : NEGBIG;
      const float mx   = redmax16(fmaxf(x0, x1));
      const float mnew = fmaxf(mrow[r], mx);
      const float p0 = __expf(x0 - mnew);
      const float p1 = __expf(x1 - mnew);
      const float rs = redsum16(p0 + p1);
      const float c  = __expf(mrow[r] - mnew);
      lrow[r] = lrow[r] * c + rs;
      mrow[r] = mnew;
      corr[r] = c;
      Pst[w][buf][half * 8 + r][l15]      = (_Float16)p0;  // D-layout -> staging
      Pst[w][buf][half * 8 + r][16 + l15] = (_Float16)p1;
    }
#pragma unroll
    for (int r = 0; r < 8; ++r) { acc0[r] *= corr[r]; acc1[r] *= corr[r]; }
    __syncthreads();  // uniform trip count; orders Pst stores vs A-frag loads
                      // (double buffer: next iteration writes the other half)

    // P A-fragment (16 queries x 32 keys) from staging
    F16x16 aP;
    { const v8h* p = (const v8h*)&Pst[w][buf][l15][half * 16];     aP.h[0] = p[0]; }
    { const v8h* p = (const v8h*)&Pst[w][buf][l15][half * 16 + 8]; aP.h[1] = p[0]; }

    // V B-fragments (lane = d column, elems = 16 consecutive keys)
    F16x16 bv0, bv1;
    { const v8h* p = (const v8h*)&VldsT[l15][kb0 + half * 16];      bv0.h[0] = p[0]; bv0.h[1] = p[1]; }
    { const v8h* p = (const v8h*)&VldsT[16 + l15][kb0 + half * 16]; bv1.h[0] = p[0]; bv1.h[1] = p[1]; }

    acc0 = __builtin_amdgcn_wmma_f32_16x16x32_f16(false, aP.v, false, bv0.v, (short)0, acc0, false, false);
    acc1 = __builtin_amdgcn_wmma_f32_16x16x32_f16(false, aP.v, false, bv1.v, (short)0, acc1, false, false);
  }

  // ---- Epilogue: out = acc / l ----
  float* op = out + ((size_t)b * SEQ + qbase) * EMB + (size_t)head * DHEAD;
#pragma unroll
  for (int r = 0; r < 8; ++r) {
    const float inv = 1.0f / lrow[r];
    op[(size_t)r * EMB + l15]      = acc0[r] * inv;
    op[(size_t)r * EMB + 16 + l15] = acc1[r] * inv;
  }
}

extern "C" void kernel_launch(void* const* d_in, const int* in_sizes, int n_in,
                              void* d_out, int out_size, void* d_ws, size_t ws_size,
                              hipStream_t stream) {
  (void)in_sizes; (void)n_in; (void)out_size; (void)d_ws; (void)ws_size;
  const float* q = (const float*)d_in[0];
  const float* k = (const float*)d_in[1];
  const float* v = (const float*)d_in[2];
  // d_in[3] = cache_index (0), unused
  float* out = (float*)d_out;
  dim3 grid(SEQ / QCHUNK, NHEADS, 4);  // (q-chunk, head, batch)
  mha_chunk_window_kernel<<<grid, 128, 0, stream>>>(q, k, v, out);
}
